// VGAE_42159398977598
// MI455X (gfx1250) — matrix operand order repaired and checked by reading, not compiled, not run on hardware
//
#include <hip/hip_runtime.h>

typedef __attribute__((ext_vector_type(16))) __bf16 v16bf;
typedef __attribute__((ext_vector_type(8)))  __bf16 v8bf;
typedef __attribute__((ext_vector_type(8)))  float  v8f;
typedef __attribute__((ext_vector_type(4)))  unsigned int u32x4;
typedef __attribute__((ext_vector_type(8)))  int i32x8;
typedef __attribute__((ext_vector_type(4)))  int i32x4;

#define HD 128   // node/hidden dim
#define H2 256   // 2*HD

#if __has_builtin(__builtin_amdgcn_tensor_load_to_lds) && \
    __has_builtin(__builtin_amdgcn_s_wait_tensorcnt)
#define HAVE_TDM 1
#else
#define HAVE_TDM 0
#endif

static __device__ __forceinline__ float warp_red_sum(float v) {
#pragma unroll
  for (int o = 16; o > 0; o >>= 1) v += __shfl_xor(v, o, 32);
  return v;
}

#if HAVE_TDM
// Low 32 bits of the flat address of a __shared__ object == LDS byte offset
// (flat LDS aperture keeps the offset in addr[31:0], ISA 10.2).
static __device__ __forceinline__ unsigned lds_off_u32(const void* p) {
  return (unsigned)(unsigned long long)(uintptr_t)p;
}

// Issue a 2D TDM tile load: tile = tileD1 rows x tileD0 elements,
// row stride strideD0 elements, element size = 1<<log2sz bytes.
// D# bitfields per CDNA5 ISA 8.3/8.4 (group0: count/lds/global/type,
// group1: data_size, tensor dims, tile dims, dim0 stride).
static __device__ __forceinline__ void tdm_load_2d(
    unsigned lds_off, const void* gptr, unsigned log2sz,
    unsigned tensorD0, unsigned tensorD1,
    unsigned tileD0, unsigned tileD1, unsigned strideD0)
{
  unsigned long long ga = (unsigned long long)(uintptr_t)gptr;
  u32x4 g0;
  g0[0] = 1u;                                             // count=1 (user D#)
  g0[1] = lds_off;                                        // lds_addr
  g0[2] = (unsigned)ga;                                   // global_addr[31:0]
  g0[3] = (unsigned)((ga >> 32) & 0x01FFFFFFu) | (2u << 30); // addr[56:32]|type=2
  i32x8 g1;
  g1[0] = (int)(log2sz << 16);                            // wg_mask=0, data_size
  g1[1] = (int)((tensorD0 & 0xFFFFu) << 16);              // tensor_dim0 lo16
  g1[2] = (int)((tensorD0 >> 16) | ((tensorD1 & 0xFFFFu) << 16));
  g1[3] = (int)((tensorD1 >> 16) | (tileD0 << 16));       // tile_dim0
  g1[4] = (int)tileD1;                                    // tile_dim1, tile_dim2=0
  g1[5] = (int)strideD0;                                  // tensor_dim0_stride lo32
  g1[6] = 0;
  g1[7] = 0;
  i32x4 z4 = {0, 0, 0, 0};
#if defined(__clang_major__) && (__clang_major__ >= 23)
  i32x8 z8 = {0, 0, 0, 0, 0, 0, 0, 0};
  __builtin_amdgcn_tensor_load_to_lds(g0, g1, z4, z4, z8, 0);
#else
  __builtin_amdgcn_tensor_load_to_lds(g0, g1, z4, z4, 0);
#endif
}
#endif  // HAVE_TDM

// ---------------------------------------------------------------------------
// GEMM1: Y[N x 256] (bf16) = bf16( Xs[N x 128] @ Wb[128 x 256] + bias )
// Block = 8 waves = 8 column tiles (grid.x=2 covers 256 cols). The 16-row A
// panel (16x128 f32, 8KB) is staged in LDS once per row block via the TDM
// (double buffered, wave 0 issues, s_wait_tensorcnt + barrier publishes) and
// shared by all 8 waves -> 8x less A traffic than register-direct loads.
// ---------------------------------------------------------------------------
__global__ __launch_bounds__(256) void gin_gemm1(
    const float* __restrict__ Xs, const __bf16* __restrict__ Wb,
    const float* __restrict__ bias, __bf16* __restrict__ Y,
    int nRB, int rbPerBlk)
{
  __shared__ float smA[2][16 * HD];      // 2 x 8KB double buffer
  const int lane = threadIdx.x & 31;
  const int wave = threadIdx.x >> 5;
  const int n    = lane & 15;
  const int hi   = lane >> 4;
  const int colBase = (blockIdx.x * 8 + wave) << 4;

  // B fragments resident in registers: K=128 -> 4 steps of 32.
  v16bf bfrag[4];
#pragma unroll
  for (int ks = 0; ks < 4; ++ks) {
    const int kO = ks * 32 + hi * 8;
#pragma unroll
    for (int i = 0; i < 16; ++i)
      bfrag[ks][i] = Wb[(kO + i + (i >= 8 ? 8 : 0)) * H2 + colBase + n];
  }
  const float bval = bias[colBase + n];

  const int rb0   = blockIdx.y * rbPerBlk;
  const int rbEnd = (rb0 + rbPerBlk < nRB) ? rb0 + rbPerBlk : nRB;
  if (rb0 >= nRB) return;  // uniform over block

#if HAVE_TDM
  if (wave == 0)
    tdm_load_2d(lds_off_u32(&smA[0][0]), Xs + (size_t)rb0 * 16 * HD,
                2u, HD, 16u, HD, 16u, HD);
#endif
  for (int rb = rb0; rb < rbEnd; ++rb) {
    const int buf = (rb - rb0) & 1;
#if HAVE_TDM
    if (wave == 0) {
      if (rb + 1 < rbEnd) {
        tdm_load_2d(lds_off_u32(&smA[buf ^ 1][0]), Xs + (size_t)(rb + 1) * 16 * HD,
                    2u, HD, 16u, HD, 16u, HD);
        __builtin_amdgcn_s_wait_tensorcnt(1);   // current panel done
      } else {
        __builtin_amdgcn_s_wait_tensorcnt(0);
      }
    }
    __syncthreads();                            // publish panel
#else
    __syncthreads();                            // protect previous compute
    {
      const float4* g4 = (const float4*)(Xs + (size_t)rb * 16 * HD);
      for (int i = threadIdx.x; i < 16 * HD / 4; i += 256)
        ((float4*)&smA[buf][0])[i] = g4[i];
    }
    __syncthreads();
#endif
    const float* arow = &smA[buf][n * HD];
    v8f acc;
#pragma unroll
    for (int i = 0; i < 8; ++i) acc[i] = bval;
#pragma unroll
    for (int ks = 0; ks < 4; ++ks) {
      const int kb = ks * 32 + hi * 8;
      float4 x0 = *(const float4*)(arow + kb);       // ds_load_b128
      float4 x1 = *(const float4*)(arow + kb + 4);
      float4 x2 = *(const float4*)(arow + kb + 16);
      float4 x3 = *(const float4*)(arow + kb + 20);
      v16bf a;
      a[0]=(__bf16)x0.x;  a[1]=(__bf16)x0.y;  a[2]=(__bf16)x0.z;  a[3]=(__bf16)x0.w;
      a[4]=(__bf16)x1.x;  a[5]=(__bf16)x1.y;  a[6]=(__bf16)x1.z;  a[7]=(__bf16)x1.w;
      a[8]=(__bf16)x2.x;  a[9]=(__bf16)x2.y;  a[10]=(__bf16)x2.z; a[11]=(__bf16)x2.w;
      a[12]=(__bf16)x3.x; a[13]=(__bf16)x3.y; a[14]=(__bf16)x3.z; a[15]=(__bf16)x3.w;
      acc = __builtin_amdgcn_wmma_f32_16x16x32_bf16(
          false, a, false, bfrag[ks], (short)0, acc, false, false);
    }
    __bf16* yout = Y + (size_t)(rb * 16 + hi * 8) * H2 + colBase + n;
#pragma unroll
    for (int v = 0; v < 8; ++v)
      yout[(size_t)v * H2] = (__bf16)acc[v];
#if HAVE_TDM
    __syncthreads();    // all waves done with this buffer before TDM reuses it
#endif
  }
}

// ---------------------------------------------------------------------------
// GEMM2: H[N x 128] (f32) = Yn[N x 256] (bf16) @ Wb[256 x 128] + bias
// Block = 8 waves = all 8 column tiles (grid.x=1). A panel 16x256 bf16 (8KB)
// via TDM double buffer -> A read exactly once from global.
// ---------------------------------------------------------------------------
__global__ __launch_bounds__(256) void gin_gemm2(
    const __bf16* __restrict__ Yn, const __bf16* __restrict__ Wb,
    const float* __restrict__ bias, float* __restrict__ Hout,
    int nRB, int rbPerBlk)
{
  __shared__ __bf16 smA[2][16 * H2];     // 2 x 8KB
  const int lane = threadIdx.x & 31;
  const int wave = threadIdx.x >> 5;
  const int n    = lane & 15;
  const int hi   = lane >> 4;
  const int colBase = wave << 4;

  v16bf bfrag[8];                        // K=256 -> 8 steps
#pragma unroll
  for (int ks = 0; ks < 8; ++ks) {
    const int kO = ks * 32 + hi * 8;
#pragma unroll
    for (int i = 0; i < 16; ++i)
      bfrag[ks][i] = Wb[(kO + i + (i >= 8 ? 8 : 0)) * HD + colBase + n];
  }
  const float bval = bias[colBase + n];

  const int rb0   = blockIdx.y * rbPerBlk;
  const int rbEnd = (rb0 + rbPerBlk < nRB) ? rb0 + rbPerBlk : nRB;
  if (rb0 >= nRB) return;

#if HAVE_TDM
  if (wave == 0)
    tdm_load_2d(lds_off_u32(&smA[0][0]), Yn + (size_t)rb0 * 16 * H2,
                1u, H2, 16u, H2, 16u, H2);
#endif
  for (int rb = rb0; rb < rbEnd; ++rb) {
    const int buf = (rb - rb0) & 1;
#if HAVE_TDM
    if (wave == 0) {
      if (rb + 1 < rbEnd) {
        tdm_load_2d(lds_off_u32(&smA[buf ^ 1][0]), Yn + (size_t)(rb + 1) * 16 * H2,
                    1u, H2, 16u, H2, 16u, H2);
        __builtin_amdgcn_s_wait_tensorcnt(1);
      } else {
        __builtin_amdgcn_s_wait_tensorcnt(0);
      }
    }
    __syncthreads();
#else
    __syncthreads();
    {
      const float4* g4 = (const float4*)(Yn + (size_t)rb * 16 * H2);
      for (int i = threadIdx.x; i < 16 * H2 * 2 / 16; i += 256)
        ((float4*)&smA[buf][0])[i] = g4[i];
    }
    __syncthreads();
#endif
    const __bf16* arow = &smA[buf][n * H2];
    v8f acc;
#pragma unroll
    for (int i = 0; i < 8; ++i) acc[i] = bval;
#pragma unroll
    for (int ks = 0; ks < 8; ++ks) {
      const int kb = ks * 32 + hi * 8;
      v8bf lo  = *(const v8bf*)(arow + kb);          // ds_load_b128
      v8bf hi8 = *(const v8bf*)(arow + kb + 16);
      v16bf a;
#pragma unroll
      for (int i = 0; i < 8; ++i) { a[i] = lo[i]; a[8 + i] = hi8[i]; }
      acc = __builtin_amdgcn_wmma_f32_16x16x32_bf16(
          false, a, false, bfrag[ks], (short)0, acc, false, false);
    }
    float* hout = Hout + (size_t)(rb * 16 + hi * 8) * HD + colBase + n;
#pragma unroll
    for (int v = 0; v < 8; ++v)
      hout[(size_t)v * HD] = acc[v];
#if HAVE_TDM
    __syncthreads();
#endif
  }
}

// ---------------------------------------------------------------------------
__global__ __launch_bounds__(256) void copy_f32v4(
    float4* __restrict__ dst, const float4* __restrict__ src, int n4)
{
  int i = blockIdx.x * blockDim.x + threadIdx.x;
  if (i < n4) dst[i] = src[i];
}

__global__ __launch_bounds__(256) void edge_scatter(
    const float* __restrict__ X, const int* __restrict__ srcI,
    const int* __restrict__ dstI, float* __restrict__ S, int E)
{
  int t = blockIdx.x * blockDim.x + threadIdx.x;
  int e = t >> 5;
  if (e >= E) return;
  int c = (t & 31) << 2;
  float4 v = *(const float4*)(X + (size_t)srcI[e] * HD + c);
  float* p = S + (size_t)dstI[e] * HD + c;
  __hip_atomic_fetch_add(p + 0, v.x, __ATOMIC_RELAXED, __HIP_MEMORY_SCOPE_AGENT);
  __hip_atomic_fetch_add(p + 1, v.y, __ATOMIC_RELAXED, __HIP_MEMORY_SCOPE_AGENT);
  __hip_atomic_fetch_add(p + 2, v.z, __ATOMIC_RELAXED, __HIP_MEMORY_SCOPE_AGENT);
  __hip_atomic_fetch_add(p + 3, v.w, __ATOMIC_RELAXED, __HIP_MEMORY_SCOPE_AGENT);
}

__global__ __launch_bounds__(256) void cvt_w_bf16(
    const float* __restrict__ W1, const float* __restrict__ W2,
    __bf16* __restrict__ O1, __bf16* __restrict__ O2, int n1, int n2)
{
  int i = blockIdx.x * blockDim.x + threadIdx.x;
  if (i < n1)            O1[i]      = (__bf16)W1[i];
  else if (i < n1 + n2)  O2[i - n1] = (__bf16)W2[i - n1];
}

__global__ void zero_f32(float* __restrict__ p, int n) {
  int i = blockIdx.x * blockDim.x + threadIdx.x;
  if (i < n) p[i] = 0.0f;
}

__global__ __launch_bounds__(256) void bn_stats(
    const __bf16* __restrict__ Y, float* __restrict__ stats,
    int rowsPer, int Nrows)
{
  int c  = threadIdx.x;
  int r0 = blockIdx.x * rowsPer;
  float s = 0.f, q = 0.f;
  for (int r = 0; r < rowsPer; ++r) {
    int row = r0 + r;
    if (row >= Nrows) break;
    float v = (float)Y[(size_t)row * H2 + c];
    s += v; q += v * v;
  }
  atomicAdd(&stats[c], s);
  atomicAdd(&stats[H2 + c], q);
}

__global__ __launch_bounds__(256) void bn_apply(
    __bf16* __restrict__ Y, const float* __restrict__ stats,
    const float* __restrict__ gamma, const float* __restrict__ beta,
    float invN, int relu, int total)
{
  int i = blockIdx.x * blockDim.x + threadIdx.x;
  if (i >= total) return;
  int c = i & (H2 - 1);
  float m   = stats[c] * invN;
  float var = stats[H2 + c] * invN - m * m;
  float v = ((float)Y[i] - m) * rsqrtf(var + 1e-5f) * gamma[c] + beta[c];
  if (relu) v = fmaxf(v, 0.f);
  Y[i] = (__bf16)v;
}

__global__ __launch_bounds__(256) void reparam_kl(
    const float* __restrict__ noise, const float* __restrict__ mu,
    const float* __restrict__ ls, float* __restrict__ z,
    float* __restrict__ kl, float scale, int total)
{
  int i = blockIdx.x * blockDim.x + threadIdx.x;
  float t = 0.f;
  if (i < total) {
    float l = ls[i], m = mu[i], e = __expf(l);
    z[i] = noise[i] * e + m;
    t = 1.f + 2.f * l - m * m - e * e;
  }
  t = warp_red_sum(t);
  if ((threadIdx.x & 31) == 0) atomicAdd(kl, t * scale);
}

__global__ __launch_bounds__(256) void node_dots(
    const float* __restrict__ z, const float* __restrict__ Wc,
    float* __restrict__ u0, float* __restrict__ u1, int Nn)
{
  int nid = blockIdx.x * blockDim.x + threadIdx.x;
  if (nid >= Nn) return;
  const float4* zr = (const float4*)(z + (size_t)nid * HD);
  const float4* w0 = (const float4*)(Wc);
  const float4* w1 = (const float4*)(Wc + HD);
  float s0 = 0.f, s1 = 0.f;
#pragma unroll
  for (int i = 0; i < HD / 4; ++i) {
    float4 a = zr[i], b = w0[i], c = w1[i];
    s0 += a.x * b.x + a.y * b.y + a.z * b.z + a.w * b.w;
    s1 += a.x * c.x + a.y * c.y + a.z * c.z + a.w * c.w;
  }
  u0[nid] = s0; u1[nid] = s1;
}

__global__ __launch_bounds__(256) void bridge_pred(
    const float* __restrict__ u0, const float* __restrict__ u1,
    const int* __restrict__ r0, const int* __restrict__ r1,
    const float* __restrict__ bc, float* __restrict__ out, int Bn)
{
  int b = blockIdx.x * blockDim.x + threadIdx.x;
  if (b >= Bn) return;
  float v = u0[r0[b]] + u1[r1[b]] + bc[0];
  out[b] = 1.0f / (1.0f + __expf(-v));
}

// ===========================================================================
extern "C" void kernel_launch(void* const* d_in, const int* in_sizes, int n_in,
                              void* d_out, int out_size, void* d_ws, size_t ws_size,
                              hipStream_t stream) {
  const float* x      = (const float*)d_in[0];
  const float* noise  = (const float*)d_in[1];
  const float* W1s    = (const float*)d_in[2];   // (4,128,256)
  const float* b1s    = (const float*)d_in[3];   // (4,256)
  const float* gammas = (const float*)d_in[4];   // (4,256)
  const float* betas  = (const float*)d_in[5];   // (4,256)
  const float* W2s    = (const float*)d_in[6];   // (4,256,128)
  const float* b2s    = (const float*)d_in[7];   // (4,128)
  const float* Wc     = (const float*)d_in[8];   // (256,)
  const float* bc     = (const float*)d_in[9];   // (1,)
  const int*   edge   = (const int*)d_in[10];    // (2,E)
  const int*   bidx   = (const int*)d_in[11];    // (2,B)

  const int N  = in_sizes[0] / HD;
  const int E  = in_sizes[10] / 2;
  const int Bp = in_sizes[11] / 2;
  const int* esrc = edge;
  const int* edst = edge + E;
  const int* br0  = bidx;
  const int* br1  = bidx + Bp;
  float* outp = (float*)d_out;          // [0..Bp) A_pred, [Bp] kl

  // ---- workspace carve ----
  char*  wsb = (char*)d_ws;
  size_t off = 0;
  auto carve = [&](size_t bytes) -> void* {
    void* p = wsb + off;
    off = (off + bytes + 255) & ~(size_t)255;
    return p;
  };
  const size_t nN = (size_t)N;
  float*  sumbuf = (float*) carve(nN * HD * sizeof(float));
  __bf16* Ybf    = (__bf16*)carve(nN * H2 * sizeof(__bf16));
  float*  hA     = (float*) carve(nN * HD * sizeof(float));
  float*  hB     = (float*) carve(nN * HD * sizeof(float));
  float*  mub    = (float*) carve(nN * HD * sizeof(float));
  float*  lsb    = (float*) carve(nN * HD * sizeof(float));
  float*  u0     = (float*) carve(nN * sizeof(float));
  float*  u1     = (float*) carve(nN * sizeof(float));
  __bf16* W1b    = (__bf16*)carve(4 * HD * H2 * sizeof(__bf16));
  __bf16* W2b    = (__bf16*)carve(4 * H2 * HD * sizeof(__bf16));
  float*  stats  = (float*) carve(2 * H2 * sizeof(float));
  float*  zbuf   = sumbuf;   // reuse: aggregation consumed before reparam
  (void)ws_size;

  // ---- config ----
  const int nRB   = N / 16;                      // 16-row blocks (4000)
  const int gy    = 500;                         // row-block grid y
  const int rbPB  = (nRB + gy - 1) / gy;         // row blocks per block (8)
  const int n4    = N * HD / 4;
  const int cpB   = (n4 + 255) / 256;
  const int scB   = (E * 32 + 255) / 256;
  const int bnB   = (N * H2 + 255) / 256;
  const int rowsPer = (N + 255) / 256;
  const float invN  = 1.0f / (float)N;
  const float klScale = 0.5f / ((float)N * (float)N);

  {
    int n1 = 4 * HD * H2, n2 = 4 * H2 * HD;
    cvt_w_bf16<<<(n1 + n2 + 255) / 256, 256, 0, stream>>>(W1s, W2s, W1b, W2b, n1, n2);
  }
  zero_f32<<<1, 32, 0, stream>>>(outp + Bp, 1);     // kl accumulator

  auto run_layer = [&](const float* Xin, int l, float* Hout, int relu, bool doAgg) {
    if (doAgg) {
      copy_f32v4<<<cpB, 256, 0, stream>>>((float4*)sumbuf, (const float4*)Xin, n4);
      edge_scatter<<<scB, 256, 0, stream>>>(Xin, esrc, edst, sumbuf, E);
    }
    gin_gemm1<<<dim3(2, gy), 256, 0, stream>>>(
        sumbuf, W1b + (size_t)l * HD * H2, b1s + l * H2, Ybf, nRB, rbPB);
    zero_f32<<<2, 256, 0, stream>>>(stats, 2 * H2);
    bn_stats<<<256, 256, 0, stream>>>(Ybf, stats, rowsPer, N);
    bn_apply<<<bnB, 256, 0, stream>>>(Ybf, stats, gammas + l * H2, betas + l * H2,
                                      invN, relu, N * H2);
    gin_gemm2<<<dim3(1, gy), 256, 0, stream>>>(
        Ybf, W2b + (size_t)l * H2 * HD, b2s + l * HD, Hout, nRB, rbPB);
  };

  run_layer(x,  0, hA,  1, true);
  run_layer(hA, 1, hB,  1, true);
  run_layer(hB, 2, mub, 0, true);    // mean head
  run_layer(hB, 3, lsb, 0, false);   // logstd head: reuse same aggregation

  reparam_kl<<<(N * HD + 255) / 256, 256, 0, stream>>>(
      noise, mub, lsb, zbuf, outp + Bp, klScale, N * HD);
  node_dots<<<(N + 255) / 256, 256, 0, stream>>>(zbuf, Wc, u0, u1, N);
  bridge_pred<<<(Bp + 255) / 256, 256, 0, stream>>>(u0, u1, br0, br1, bc, outp, Bp);
}
